// TransformerBlock_73650099192379
// MI455X (gfx1250) — matrix operand order repaired and checked by reading, not compiled, not run on hardware
//
#include <hip/hip_runtime.h>
#include <hip/hip_bf16.h>

// ---------------- problem constants ----------------
#define B_   2
#define T_   2048
#define C_   1024
#define H_   16
#define DH_  64
#define FF_  4096
#define M_   (B_ * T_)          // 4096 rows of activations

typedef __attribute__((ext_vector_type(16))) __bf16 v16bf;
typedef __attribute__((ext_vector_type(8)))  float  v8f;

union FragBf {
    v16bf v;
    unsigned short u[16];
};

static __device__ __forceinline__ unsigned short f32_to_bf16(float f) {
    unsigned int x = __float_as_uint(f);
    unsigned int r = x + 0x7FFFu + ((x >> 16) & 1u);   // round-to-nearest-even
    return (unsigned short)(r >> 16);
}

static __device__ __forceinline__ v8f zero8() {
    v8f z;
#pragma unroll
    for (int i = 0; i < 8; ++i) z[i] = 0.0f;
    return z;
}

// ---------------- weight prep kernels ----------------

// wq/wk/wv: [H][C][DH] fp32  ->  [C][H*DH] bf16  (head-concat column layout)
__global__ __launch_bounds__(256)
void k_qkv_transpose(const float* __restrict__ w, unsigned short* __restrict__ out) {
    int idx = blockIdx.x * 256 + threadIdx.x;          // < H_*C_*DH_ = 1M
    int d = idx & 63;
    int c = (idx >> 6) & 1023;
    int h = idx >> 16;
    out[(size_t)c * C_ + h * DH_ + d] = f32_to_bf16(w[idx]);
}

__global__ __launch_bounds__(256)
void k_f32_to_bf16(const float* __restrict__ src, unsigned short* __restrict__ dst, int n) {
    int idx = blockIdx.x * 256 + threadIdx.x;
    if (idx < n) dst[idx] = f32_to_bf16(src[idx]);
}

// ---------------- layernorm (fp32 in -> bf16 out) ----------------
__global__ __launch_bounds__(256)
void k_layernorm_bf16(const float* __restrict__ x, const float* __restrict__ g,
                      const float* __restrict__ b, unsigned short* __restrict__ out) {
    __shared__ float s1[8], s2[8];
    const int row = blockIdx.x;
    const float* xr = x + (size_t)row * C_;
    float4 v4 = ((const float4*)xr)[threadIdx.x];
    float a = v4.x + v4.y + v4.z + v4.w;
    float q = v4.x * v4.x + v4.y * v4.y + v4.z * v4.z + v4.w * v4.w;
#pragma unroll
    for (int o = 16; o > 0; o >>= 1) {
        a += __shfl_xor(a, o, 32);
        q += __shfl_xor(q, o, 32);
    }
    int wv = threadIdx.x >> 5, ln = threadIdx.x & 31;
    if (ln == 0) { s1[wv] = a; s2[wv] = q; }
    __syncthreads();
    a = 0.f; q = 0.f;
#pragma unroll
    for (int i = 0; i < 8; ++i) { a += s1[i]; q += s2[i]; }
    float mu   = a * (1.0f / C_);
    float var  = q * (1.0f / C_) - mu * mu;
    float rstd = rsqrtf(var + 1e-5f);
    int c0 = threadIdx.x * 4;
    unsigned short* orow = out + (size_t)row * C_;
    float vals[4] = {v4.x, v4.y, v4.z, v4.w};
#pragma unroll
    for (int i = 0; i < 4; ++i)
        orow[c0 + i] = f32_to_bf16((vals[i] - mu) * rstd * g[c0 + i] + b[c0 + i]);
}

// ---------------- WMMA bf16 GEMM: out = act(A @ W + bias) [+ resid] ----------------
// A   : [M][K] bf16 row-major
// W   : [K][N] bf16 row-major
// tile: block 128x64, K-step 32; 8 waves as 4(M) x 2(N), each wave 32x32 (2x2 WMMA)
#define GBM 128
#define GBN 64
#define GBK 32
#define LDA 34   // 32 + 2 pad (shorts)
#define LDB 34

__global__ __launch_bounds__(256)
void k_gemm_bf16(const unsigned short* __restrict__ A,
                 const unsigned short* __restrict__ W,
                 const float* __restrict__ bias,
                 const float* __restrict__ resid,
                 float* __restrict__ outF,
                 unsigned short* __restrict__ outB,
                 int M, int N, int K, int relu) {
    __shared__ unsigned short As[GBM * LDA];
    __shared__ unsigned short Bs[GBN * LDB];   // stored transposed: Bs[n][k]

    const int tid  = threadIdx.x;
    const int wave = tid >> 5;
    const int lane = tid & 31;
    const int half = lane >> 4;
    const int l16  = lane & 15;
    const int m0   = blockIdx.y * GBM;
    const int n0   = blockIdx.x * GBN;
    const int wm   = (wave & 3) * 32;
    const int wn   = (wave >> 2) * 32;

    v8f acc[2][2];
#pragma unroll
    for (int i = 0; i < 2; ++i)
#pragma unroll
        for (int j = 0; j < 2; ++j) acc[i][j] = zero8();

    for (int k0 = 0; k0 < K; k0 += GBK) {
        __syncthreads();
        // --- stage A tile: 128x32, each thread 16 shorts ---
        {
            int row = tid >> 1, seg = tid & 1;
            const uint4* pa = (const uint4*)(A + (size_t)(m0 + row) * K + k0 + seg * 16);
            uint4 a0 = pa[0], a1 = pa[1];
            unsigned int* da = (unsigned int*)&As[row * LDA + seg * 16];
            da[0] = a0.x; da[1] = a0.y; da[2] = a0.z; da[3] = a0.w;
            da[4] = a1.x; da[5] = a1.y; da[6] = a1.z; da[7] = a1.w;
            if (k0 + GBK < K)
                __builtin_prefetch(A + (size_t)(m0 + row) * K + k0 + GBK + seg * 16, 0, 0);
        }
        // --- stage B tile 32x64, store transposed [n][k] ---
        {
            int kr = tid >> 3, seg = tid & 7;          // kr: k row, seg: 8-col group
            uint4 wv4 = *(const uint4*)(W + (size_t)(k0 + kr) * N + n0 + seg * 8);
            unsigned int tmp[4] = {wv4.x, wv4.y, wv4.z, wv4.w};
#pragma unroll
            for (int j = 0; j < 4; ++j) {
                Bs[(seg * 8 + 2 * j)     * LDB + kr] = (unsigned short)(tmp[j] & 0xffffu);
                Bs[(seg * 8 + 2 * j + 1) * LDB + kr] = (unsigned short)(tmp[j] >> 16);
            }
            if (k0 + GBK < K)
                __builtin_prefetch(W + (size_t)(k0 + GBK + kr) * N + n0 + seg * 8, 0, 0);
        }
        __syncthreads();

        // --- gather fragments (documented CDNA5 16-bit layouts) ---
        FragBf af[2], bf[2];
#pragma unroll
        for (int i = 0; i < 2; ++i) {
            int row = wm + i * 16 + l16;
#pragma unroll
            for (int e = 0; e < 16; e += 2) {
                int kk = ((e & 8) << 1) + 8 * half + (e & 7);
                unsigned int w = *(const unsigned int*)&As[row * LDA + kk];
                af[i].u[e] = (unsigned short)(w & 0xffffu);
                af[i].u[e + 1] = (unsigned short)(w >> 16);
            }
        }
#pragma unroll
        for (int j = 0; j < 2; ++j) {
            int col = wn + j * 16 + l16;
#pragma unroll
            for (int e = 0; e < 16; e += 2) {
                int kk = ((e & 8) << 1) + 8 * half + (e & 7);
                unsigned int w = *(const unsigned int*)&Bs[col * LDB + kk];
                bf[j].u[e] = (unsigned short)(w & 0xffffu);
                bf[j].u[e + 1] = (unsigned short)(w >> 16);
            }
        }
#pragma unroll
        for (int i = 0; i < 2; ++i)
#pragma unroll
            for (int j = 0; j < 2; ++j)
                acc[i][j] = __builtin_amdgcn_wmma_f32_16x16x32_bf16(
                    false, af[i].v, false, bf[j].v, (short)0, acc[i][j], false, false);
    }

    // --- epilogue: bias / relu / residual / store ---
#pragma unroll
    for (int i = 0; i < 2; ++i)
#pragma unroll
        for (int j = 0; j < 2; ++j) {
            int col = n0 + wn + j * 16 + l16;
            float bv = bias ? bias[col] : 0.0f;
#pragma unroll
            for (int r = 0; r < 8; ++r) {
                int row = m0 + wm + i * 16 + r + 8 * half;
                float v = acc[i][j][r] + bv;
                if (relu) v = fmaxf(v, 0.0f);
                size_t off = (size_t)row * N + col;
                if (resid) v += resid[off];
                if (outF) outF[off] = v;
                if (outB) outB[off] = f32_to_bf16(v);
            }
        }
}

// ---------------- flash attention (bf16 QKV, fp32 softmax/accum) ----------------
// q/k/v/out layout: [B*T][C] with column h*DH+d (head-concat). grid: (T/64, B*H), 128 thr.
#define LKS 66   // K tile row stride (64 + 2)
#define LVS 34   // V^T tile row stride (32 + 2)

__global__ __launch_bounds__(128)
void k_attention(const unsigned short* __restrict__ qg,
                 const unsigned short* __restrict__ kg,
                 const unsigned short* __restrict__ vg,
                 unsigned short* __restrict__ og) {
    __shared__ unsigned short Ks[32 * LKS];        // [key][d]
    __shared__ unsigned short Vs[64 * LVS];        // [d][key]  (transposed)
    __shared__ unsigned short Ps[4][16 * LVS];     // per-wave P tile [row][key]

    const int bh   = blockIdx.y;
    const int b    = bh >> 4;
    const int h    = bh & 15;
    const int q0   = blockIdx.x * 64;
    const int wave = threadIdx.x >> 5;
    const int lane = threadIdx.x & 31;
    const int half = lane >> 4;
    const int l16  = lane & 15;
    const int qt   = q0 + wave * 16;               // this wave's first query row
    const size_t base = (size_t)b * T_ * C_ + (size_t)h * DH_;

    // preload Q fragments (2 K-chunks of 32 over DH=64)
    FragBf qf[2];
#pragma unroll
    for (int c = 0; c < 2; ++c) {
        const unsigned short* qr = qg + base + (size_t)(qt + l16) * C_ + c * 32;
#pragma unroll
        for (int e = 0; e < 16; e += 2) {
            int kk = ((e & 8) << 1) + 8 * half + (e & 7);
            unsigned int w = *(const unsigned int*)(qr + kk);
            qf[c].u[e] = (unsigned short)(w & 0xffffu);
            qf[c].u[e + 1] = (unsigned short)(w >> 16);
        }
    }

    float mrow[8], lrow[8];
    v8f acc[4];
#pragma unroll
    for (int r = 0; r < 8; ++r) { mrow[r] = -1e30f; lrow[r] = 0.0f; }
#pragma unroll
    for (int n = 0; n < 4; ++n) acc[n] = zero8();

    const int nkb = (q0 + 64) >> 5;                // causal: key blocks for this CTA
    for (int kb = 0; kb < nkb; ++kb) {
        const int s0 = kb * 32;
        __syncthreads();
        // stage K (row-major) and V (transposed): 32 keys x 64 d, 16 shorts/thread
        {
            int kr = threadIdx.x >> 2, seg = threadIdx.x & 3;
            const unsigned short* src = kg + base + (size_t)(s0 + kr) * C_ + seg * 16;
            const uint4* pk = (const uint4*)src;
            uint4 k0v = pk[0], k1v = pk[1];
            unsigned int* dk = (unsigned int*)&Ks[kr * LKS + seg * 16];
            dk[0] = k0v.x; dk[1] = k0v.y; dk[2] = k0v.z; dk[3] = k0v.w;
            dk[4] = k1v.x; dk[5] = k1v.y; dk[6] = k1v.z; dk[7] = k1v.w;

            const uint4* pv = (const uint4*)(vg + base + (size_t)(s0 + kr) * C_ + seg * 16);
            uint4 v0 = pv[0], v1 = pv[1];
            unsigned int tmp[8] = {v0.x, v0.y, v0.z, v0.w, v1.x, v1.y, v1.z, v1.w};
#pragma unroll
            for (int j = 0; j < 8; ++j) {
                Vs[(seg * 16 + 2 * j)     * LVS + kr] = (unsigned short)(tmp[j] & 0xffffu);
                Vs[(seg * 16 + 2 * j + 1) * LVS + kr] = (unsigned short)(tmp[j] >> 16);
            }
        }
        __syncthreads();

        if (s0 <= qt + 15) {                       // wave-uniform causal skip
            // scores S[16][32]: two 16x16 tiles, each accumulated over 2 K-chunks
            v8f sf[2];
#pragma unroll
            for (int kt = 0; kt < 2; ++kt) {
                v8f c = zero8();
#pragma unroll
                for (int ch = 0; ch < 2; ++ch) {
                    FragBf kf;
                    int n = kt * 16 + l16;
#pragma unroll
                    for (int e = 0; e < 16; e += 2) {
                        int kk = ch * 32 + ((e & 8) << 1) + 8 * half + (e & 7);
                        unsigned int w = *(const unsigned int*)&Ks[n * LKS + kk];
                        kf.u[e] = (unsigned short)(w & 0xffffu);
                        kf.u[e + 1] = (unsigned short)(w >> 16);
                    }
                    c = __builtin_amdgcn_wmma_f32_16x16x32_bf16(
                        false, qf[ch].v, false, kf.v, (short)0, c, false, false);
                }
                sf[kt] = c;
            }
            // online softmax (rows r+8*half live across the 16 lanes of this half)
#pragma unroll
            for (int r = 0; r < 8; ++r) {
                int trow = qt + r + 8 * half;
                float v0 = sf[0][r] * 0.125f;      // 1/sqrt(64)
                float v1 = sf[1][r] * 0.125f;
                if (s0 + l16 > trow)      v0 = -1e30f;
                if (s0 + 16 + l16 > trow) v1 = -1e30f;
                float mx = fmaxf(v0, v1);
#pragma unroll
                for (int o = 8; o > 0; o >>= 1) mx = fmaxf(mx, __shfl_xor(mx, o, 32));
                float mnew  = fmaxf(mrow[r], mx);
                float scale = __expf(mrow[r] - mnew);
                mrow[r] = mnew;
                float p0 = (v0 <= -1e29f) ? 0.0f : __expf(v0 - mnew);
                float p1 = (v1 <= -1e29f) ? 0.0f : __expf(v1 - mnew);
                float rs = p0 + p1;
#pragma unroll
                for (int o = 8; o > 0; o >>= 1) rs += __shfl_xor(rs, o, 32);
                lrow[r] = lrow[r] * scale + rs;
#pragma unroll
                for (int nt = 0; nt < 4; ++nt) acc[nt][r] *= scale;
                int row = r + 8 * half;
                Ps[wave][row * LVS + l16]      = f32_to_bf16(p0);
                Ps[wave][row * LVS + 16 + l16] = f32_to_bf16(p1);
            }
            // P (16x32) as A-fragment, then O += P @ V over 4 d-tiles
            FragBf pf;
#pragma unroll
            for (int e = 0; e < 16; e += 2) {
                int kk = ((e & 8) << 1) + 8 * half + (e & 7);
                unsigned int w = *(const unsigned int*)&Ps[wave][l16 * LVS + kk];
                pf.u[e] = (unsigned short)(w & 0xffffu);
                pf.u[e + 1] = (unsigned short)(w >> 16);
            }
#pragma unroll
            for (int nt = 0; nt < 4; ++nt) {
                FragBf vf;
#pragma unroll
                for (int e = 0; e < 16; e += 2) {
                    int kk = ((e & 8) << 1) + 8 * half + (e & 7);
                    unsigned int w = *(const unsigned int*)&Vs[(nt * 16 + l16) * LVS + kk];
                    vf.u[e] = (unsigned short)(w & 0xffffu);
                    vf.u[e + 1] = (unsigned short)(w >> 16);
                }
                acc[nt] = __builtin_amdgcn_wmma_f32_16x16x32_bf16(
                    false, pf.v, false, vf.v, (short)0, acc[nt], false, false);
            }
        }
    }

    // normalize and store bf16 output in head-concat layout
#pragma unroll
    for (int r = 0; r < 8; ++r) {
        float inv = 1.0f / lrow[r];
        int t = qt + r + 8 * half;
#pragma unroll
        for (int nt = 0; nt < 4; ++nt) {
            size_t off = base + (size_t)t * C_ + nt * 16 + l16;
            og[off] = f32_to_bf16(acc[nt][r] * inv);
        }
    }
}

// ---------------- host orchestration ----------------
extern "C" void kernel_launch(void* const* d_in, const int* in_sizes, int n_in,
                              void* d_out, int out_size, void* d_ws, size_t ws_size,
                              hipStream_t stream) {
    (void)in_sizes; (void)n_in; (void)out_size; (void)ws_size;
    const float* x      = (const float*)d_in[0];
    const float* wq     = (const float*)d_in[1];
    const float* wk     = (const float*)d_in[2];
    const float* wv     = (const float*)d_in[3];
    const float* w_proj = (const float*)d_in[4];
    const float* b_proj = (const float*)d_in[5];
    const float* g_ln1  = (const float*)d_in[6];
    const float* b_ln1  = (const float*)d_in[7];
    const float* g_ln2  = (const float*)d_in[8];
    const float* b_ln2  = (const float*)d_in[9];
    const float* w_ff1  = (const float*)d_in[10];
    const float* b_ff1  = (const float*)d_in[11];
    const float* w_ff2  = (const float*)d_in[12];
    const float* b_ff2  = (const float*)d_in[13];
    float* out = (float*)d_out;

    char* ws = (char*)d_ws;
    size_t o = 0;
    auto alloc = [&](size_t bytes) -> void* {
        void* p = ws + o;
        o = (o + bytes + 255) & ~(size_t)255;
        return p;
    };
    // persistent: bf16 weights (24MB) + x2 (16MB)
    unsigned short* wqt  = (unsigned short*)alloc((size_t)C_ * C_ * 2);
    unsigned short* wkt  = (unsigned short*)alloc((size_t)C_ * C_ * 2);
    unsigned short* wvt  = (unsigned short*)alloc((size_t)C_ * C_ * 2);
    unsigned short* wpb  = (unsigned short*)alloc((size_t)C_ * C_ * 2);
    unsigned short* wf1b = (unsigned short*)alloc((size_t)C_ * FF_ * 2);
    unsigned short* wf2b = (unsigned short*)alloc((size_t)FF_ * C_ * 2);
    float*          x2   = (float*)alloc((size_t)M_ * C_ * 4);
    // attn output region, later reused for h2 (both 8MB, non-overlapping lifetimes)
    unsigned short* attnb = (unsigned short*)alloc((size_t)M_ * C_ * 2);
    unsigned short* h2b   = attnb;
    // 32MB region: h1 + q + k + v, later reused for the 4096-wide FF activation
    unsigned short* h1b = (unsigned short*)alloc((size_t)M_ * C_ * 2);
    unsigned short* qb  = (unsigned short*)alloc((size_t)M_ * C_ * 2);
    unsigned short* kb  = (unsigned short*)alloc((size_t)M_ * C_ * 2);
    unsigned short* vb  = (unsigned short*)alloc((size_t)M_ * C_ * 2);
    unsigned short* ffb = h1b;   // M_*FF_ bf16 = 32MB overlays h1/q/k/v

    // 1) weight prep
    k_qkv_transpose<<<(H_ * C_ * DH_) / 256, 256, 0, stream>>>(wq, wqt);
    k_qkv_transpose<<<(H_ * C_ * DH_) / 256, 256, 0, stream>>>(wk, wkt);
    k_qkv_transpose<<<(H_ * C_ * DH_) / 256, 256, 0, stream>>>(wv, wvt);
    k_f32_to_bf16<<<(C_ * C_) / 256, 256, 0, stream>>>(w_proj, wpb, C_ * C_);
    k_f32_to_bf16<<<(C_ * FF_) / 256, 256, 0, stream>>>(w_ff1, wf1b, C_ * FF_);
    k_f32_to_bf16<<<(FF_ * C_) / 256, 256, 0, stream>>>(w_ff2, wf2b, FF_ * C_);

    // 2) LN1
    k_layernorm_bf16<<<M_, 256, 0, stream>>>(x, g_ln1, b_ln1, h1b);

    // 3) QKV projections (head-concat columns)
    dim3 gC(C_ / GBN, M_ / GBM);
    k_gemm_bf16<<<gC, 256, 0, stream>>>(h1b, wqt, nullptr, nullptr, nullptr, qb, M_, C_, C_, 0);
    k_gemm_bf16<<<gC, 256, 0, stream>>>(h1b, wkt, nullptr, nullptr, nullptr, kb, M_, C_, C_, 0);
    k_gemm_bf16<<<gC, 256, 0, stream>>>(h1b, wvt, nullptr, nullptr, nullptr, vb, M_, C_, C_, 0);

    // 4) causal attention
    k_attention<<<dim3(T_ / 64, B_ * H_), 128, 0, stream>>>(qb, kb, vb, attnb);

    // 5) output projection + first residual -> x2 (fp32)
    k_gemm_bf16<<<gC, 256, 0, stream>>>(attnb, wpb, b_proj, x, x2, nullptr, M_, C_, C_, 0);

    // 6) LN2 (h2 reuses the attn buffer)
    k_layernorm_bf16<<<M_, 256, 0, stream>>>(x2, g_ln2, b_ln2, h2b);

    // 7) FF1 + ReLU -> ffb (bf16, 4096 wide)
    dim3 gF(FF_ / GBN, M_ / GBM);
    k_gemm_bf16<<<gF, 256, 0, stream>>>(h2b, wf1b, b_ff1, nullptr, nullptr, ffb, M_, FF_, C_, 1);

    // 8) FF2 + second residual -> final fp32 output
    k_gemm_bf16<<<gC, 256, 0, stream>>>(ffb, wf2b, b_ff2, x2, out, nullptr, M_, C_, FF_, 0);
}
